// LSTMSimple_14989435863274
// MI455X (gfx1250) — matrix operand order repaired and checked by reading, not compiled, not run on hardware
//
#include <hip/hip_runtime.h>
#include <hip/hip_fp16.h>

// ---------------------------------------------------------------------------
// LSTM (B=64,S=512,D=H=O=1024) for gfx1250 / MI455X
// f16 WMMA (v_wmma_f32_16x16x32_f16) with f32 accumulation.
//
// PERSISTENT kernel: 64 blocks x 256 threads launched ONCE; the 512-step
// recurrence runs inside the kernel with a device-wide generation barrier
// (atomic arrive + s_sleep spin) instead of 512 sequential launches --
// per-step compute is ~1.2 GFLOP (sub-microsecond at f16 WMMA rates), so
// dispatch latency, not math or HBM, is the bottleneck this removes.
//
// Per step:
//   waves 0..255  : 4-gate GEMM over K=2048 ([x_t | h_{t-1}], weights L2-
//                   resident) + in-register sigmoid/tanh cell update
//                   (c in place; h double-buffered; f16 h for next GEMM)
//   waves 256..511: projection y_{t-1} = h_{t-1} @ Wp^T + bp
// ---------------------------------------------------------------------------

typedef _Float16 half16 __attribute__((ext_vector_type(16)));
typedef _Float16 half8  __attribute__((ext_vector_type(8)));
typedef _Float16 half4  __attribute__((ext_vector_type(4)));
typedef float    f32x8  __attribute__((ext_vector_type(8)));

#define WMMA_F16(a, b, c) \
    __builtin_amdgcn_wmma_f32_16x16x32_f16(false, (a), false, (b), (short)0, (c), false, false)

constexpr int B = 64, S = 512, D = 1024, H = 1024, O = 1024;
constexpr int NG = 4 * H;          // 4096 gate outputs
constexpr int KC = D + H;          // 2048 concatenated reduction dim
constexpr unsigned NBLOCKS = 64;   // persistent grid

__device__ __forceinline__ float sigf(float x) {
    return 1.0f / (1.0f + __expf(-x));
}
__device__ __forceinline__ float tanh_fast(float x) {
    float e = __expf(2.0f * x);
    return (e - 1.0f) / (e + 1.0f);
}

// Two contiguous 16-byte runs -> one 16x32 f16 fragment's per-lane 8 VGPRs.
__device__ __forceinline__ half16 load_frag2(const _Float16* p0, const _Float16* p1) {
    union { half16 v; half8 h[2]; } u;
    u.h[0] = *(const half8*)p0;
    u.h[1] = *(const half8*)p1;
    return u.v;
}

// ---------------------------------------------------------------------------
// Device-wide generation barrier (all NBLOCKS blocks must be resident; 64
// blocks x 8 waves is far under residency limits). Release fence before
// arrive, acquire on the generation flip, s_sleep while spinning.
// ---------------------------------------------------------------------------
__device__ __forceinline__ void grid_sync(unsigned* cnt, unsigned* gen) {
    __threadfence();   // release this block's stores to agent scope
    __syncthreads();
    if (threadIdx.x == 0) {
        unsigned g = __hip_atomic_load(gen, __ATOMIC_RELAXED, __HIP_MEMORY_SCOPE_AGENT);
        unsigned arrived =
            __hip_atomic_fetch_add(cnt, 1u, __ATOMIC_ACQ_REL, __HIP_MEMORY_SCOPE_AGENT) + 1u;
        if (arrived == NBLOCKS) {
            __hip_atomic_store(cnt, 0u, __ATOMIC_RELAXED, __HIP_MEMORY_SCOPE_AGENT);
            __hip_atomic_store(gen, g + 1u, __ATOMIC_RELEASE, __HIP_MEMORY_SCOPE_AGENT);
        } else {
            while (__hip_atomic_load(gen, __ATOMIC_ACQUIRE, __HIP_MEMORY_SCOPE_AGENT) == g) {
                __builtin_amdgcn_s_sleep(2);
            }
        }
    }
    __syncthreads();
    __threadfence();   // acquire remote stores
}

// ---------------------------------------------------------------------------
// Gate GEMM + cell update for one (m-tile, k-tile): 4 gate accumulators.
// A layout (16x32 f16): lane(lo) = row m, runs at kb+hi*16 and kb+hi*16+8.
// B layout (32x16 f16): lane(lo) = col n, runs at kb+hi*16 and kb+hi*16+8.
// C/D layout: element r of lane -> (m = m0+r+8*hi, n = k0+lo).
// ---------------------------------------------------------------------------
__device__ __forceinline__ void gate_wave(
    int job, int t,
    const _Float16* __restrict__ x16,
    const _Float16* __restrict__ Wcat,
    const float*    __restrict__ bsum,
    float*          __restrict__ cbuf,
    const _Float16* __restrict__ h16p,
    _Float16*       __restrict__ h16n,
    float*          __restrict__ hf32n)
{
    const int lane = threadIdx.x & 31;
    const int lo = lane & 15, hi = lane >> 4;
    const int m0 = (job >> 6) << 4;   // 4 m-tiles cover B=64
    const int k0 = (job & 63) << 4;   // 64 k-tiles cover H=1024

    const int ab = m0 + lo;           // this lane's A row (batch index)
    const _Float16* xrow = x16 + ((size_t)ab * S + t) * D;
    const _Float16* hrow = h16p + (size_t)ab * H;

    const _Float16* wc0 = Wcat + (size_t)(0 * H + k0 + lo) * KC;
    const _Float16* wc1 = Wcat + (size_t)(1 * H + k0 + lo) * KC;
    const _Float16* wc2 = Wcat + (size_t)(2 * H + k0 + lo) * KC;
    const _Float16* wc3 = Wcat + (size_t)(3 * H + k0 + lo) * KC;

    f32x8 ai = {}, af = {}, ao = {}, ac = {};

    // Phase 1: K in [0,D) from x_t ; Phase 2: K in [D,KC) from h_{t-1}.
    for (int kb = 0; kb < D; kb += 32) {
        const _Float16* ap = xrow + kb;
        half16 a = load_frag2(ap + hi * 16, ap + hi * 16 + 8);
        __builtin_prefetch(wc0 + kb + 64, 0, 0);   // global_prefetch_b8
        half16 b0 = load_frag2(wc0 + kb + hi * 16, wc0 + kb + hi * 16 + 8);
        ai = WMMA_F16(a, b0, ai);
        half16 b1 = load_frag2(wc1 + kb + hi * 16, wc1 + kb + hi * 16 + 8);
        af = WMMA_F16(a, b1, af);
        half16 b2 = load_frag2(wc2 + kb + hi * 16, wc2 + kb + hi * 16 + 8);
        ao = WMMA_F16(a, b2, ao);
        half16 b3 = load_frag2(wc3 + kb + hi * 16, wc3 + kb + hi * 16 + 8);
        ac = WMMA_F16(a, b3, ac);
    }
    for (int kb = D; kb < KC; kb += 32) {
        const _Float16* ap = hrow + (kb - D);
        half16 a = load_frag2(ap + hi * 16, ap + hi * 16 + 8);
        half16 b0 = load_frag2(wc0 + kb + hi * 16, wc0 + kb + hi * 16 + 8);
        ai = WMMA_F16(a, b0, ai);
        half16 b1 = load_frag2(wc1 + kb + hi * 16, wc1 + kb + hi * 16 + 8);
        af = WMMA_F16(a, b1, af);
        half16 b2 = load_frag2(wc2 + kb + hi * 16, wc2 + kb + hi * 16 + 8);
        ao = WMMA_F16(a, b2, ao);
        half16 b3 = load_frag2(wc3 + kb + hi * 16, wc3 + kb + hi * 16 + 8);
        ac = WMMA_F16(a, b3, ac);
    }

    // Pointwise cell update, in-register on this wave's tile.
    const int k = k0 + lo;
    const float bi = bsum[0 * H + k];
    const float bf = bsum[1 * H + k];
    const float bo = bsum[2 * H + k];
    const float bc = bsum[3 * H + k];

#pragma unroll
    for (int r = 0; r < 8; ++r) {
        const int bb = m0 + r + 8 * hi;          // batch row of this element
        const size_t idx = (size_t)bb * H + k;
        const float gi = sigf(ai[r] + bi);
        const float gf = sigf(af[r] + bf);
        const float go = sigf(ao[r] + bo);
        const float gc = tanh_fast(ac[r] + bc);
        const float cn = gf * cbuf[idx] + gi * gc;
        const float hn = go * tanh_fast(cn);
        cbuf[idx]  = cn;
        hf32n[idx] = hn;
        h16n[idx]  = (_Float16)hn;
    }
}

// ---------------------------------------------------------------------------
// Projection y_{tprev} = h @ Wp^T + bp for one 16x16 output tile.
// ---------------------------------------------------------------------------
__device__ __forceinline__ void proj_wave(
    int job, int tprev,
    const _Float16* __restrict__ h16p,
    const _Float16* __restrict__ Wp16,
    const float*    __restrict__ bp,
    float*          __restrict__ out)
{
    const int lane = threadIdx.x & 31;
    const int lo = lane & 15, hi = lane >> 4;
    const int m0 = (job >> 6) << 4;   // 4 m-tiles
    const int n0 = (job & 63) << 4;   // 64 n-tiles cover O=1024

    const _Float16* hrow = h16p + (size_t)(m0 + lo) * H;
    const _Float16* wcol = Wp16 + (size_t)(n0 + lo) * H;

    f32x8 acc = {};
    for (int kb = 0; kb < H; kb += 32) {
        half16 a = load_frag2(hrow + kb + hi * 16, hrow + kb + hi * 16 + 8);
        half16 b = load_frag2(wcol + kb + hi * 16, wcol + kb + hi * 16 + 8);
        acc = WMMA_F16(a, b, acc);
    }

    const float bias = bp[n0 + lo];
#pragma unroll
    for (int r = 0; r < 8; ++r) {
        const int bb = m0 + r + 8 * hi;
        out[((size_t)bb * S + tprev) * O + n0 + lo] = acc[r] + bias;
    }
}

// ---------------------------------------------------------------------------
// Persistent recurrence kernel: whole scan + all projections in one launch.
// ---------------------------------------------------------------------------
__global__ __launch_bounds__(256) void k_lstm_persistent(
    const _Float16* __restrict__ x16,
    const _Float16* __restrict__ Wcat,
    const _Float16* __restrict__ Wp16,
    const float*    __restrict__ bsum,
    const float*    __restrict__ bp,
    float*          __restrict__ cbuf,
    _Float16*       __restrict__ h16a,
    _Float16*       __restrict__ h16b,
    float*          __restrict__ h32a,
    float*          __restrict__ h32b,
    float*          __restrict__ out,
    unsigned*       __restrict__ bar_cnt,
    unsigned*       __restrict__ bar_gen)
{
    const int w = blockIdx.x * 8 + (threadIdx.x >> 5);

    for (int t = 0; t < S; ++t) {
        // h_{t-1} lives in buffer (t&1==0 ? a : b); h_t written to the other.
        const _Float16* hp = (t & 1) ? h16b : h16a;
        _Float16*       hn = (t & 1) ? h16a : h16b;
        float*        h32n = (t & 1) ? h32a : h32b;

        if (w < 256) {
            gate_wave(w, t, x16, Wcat, bsum, cbuf, hp, hn, h32n);
        } else if (t > 0) {
            proj_wave(w - 256, t - 1, hp, Wp16, bp, out);
        }
        grid_sync(bar_cnt, bar_gen);
    }

    // Final projection: h_{S-1} is in buffer a (S even).
    if (w >= 256) {
        proj_wave(w - 256, S - 1, h16a, Wp16, bp, out);
    }
}

// ---------------------------------------------------------------------------
// Setup / epilogue kernels
// ---------------------------------------------------------------------------
__global__ __launch_bounds__(256) void k_convert_x(
    const float* __restrict__ x, _Float16* __restrict__ x16)
{
    const size_t i = ((size_t)blockIdx.x * 256 + threadIdx.x) * 4;
    const float4 v = *(const float4*)(x + i);
    half4 o = { (_Float16)v.x, (_Float16)v.y, (_Float16)v.z, (_Float16)v.w };
    *(half4*)(x16 + i) = o;
}

// Wcat[n][k] : n = g*H + kk ; k<D -> Wx[g][kk][k], else Wh[g][kk][k-D].
__global__ __launch_bounds__(256) void k_build_wcat(
    const float* __restrict__ Wx, const float* __restrict__ Wh,
    const float* __restrict__ bx, const float* __restrict__ bh,
    _Float16* __restrict__ Wcat, float* __restrict__ bsum)
{
    const size_t i = (size_t)blockIdx.x * 256 + threadIdx.x;
    const size_t nrow = i >> 11;
    const int k = (int)(i & 2047);
    const float v = (k < D) ? Wx[nrow * D + k] : Wh[nrow * H + (k - D)];
    Wcat[i] = (_Float16)v;
    if (i < (size_t)NG) bsum[i] = bx[i] + bh[i];
}

__global__ __launch_bounds__(256) void k_build_wp(
    const float* __restrict__ Wp, _Float16* __restrict__ Wp16)
{
    const size_t i = (size_t)blockIdx.x * 256 + threadIdx.x;
    Wp16[i] = (_Float16)Wp[i];
}

__global__ __launch_bounds__(256) void k_init_state(
    float* __restrict__ c, float* __restrict__ h32a, float* __restrict__ h32b,
    _Float16* __restrict__ h16a, _Float16* __restrict__ h16b,
    unsigned* __restrict__ bar_cnt, unsigned* __restrict__ bar_gen, int n)
{
    const int i = blockIdx.x * 256 + threadIdx.x;
    if (i < n) {
        c[i] = 0.0f; h32a[i] = 0.0f; h32b[i] = 0.0f;
        h16a[i] = (_Float16)0.0f; h16b[i] = (_Float16)0.0f;
    }
    if (i == 0) { *bar_cnt = 0u; *bar_gen = 0u; }   // ws is poisoned; re-arm barrier
}

__global__ __launch_bounds__(256) void k_tail(
    const float* __restrict__ cbuf, const float* __restrict__ hfin,
    float* __restrict__ out_tail, int n)
{
    const int i = blockIdx.x * 256 + threadIdx.x;
    if (i < n) {
        out_tail[i]     = cbuf[i];
        out_tail[n + i] = hfin[i];
    }
}

// ---------------------------------------------------------------------------
// Launch
// ---------------------------------------------------------------------------
extern "C" void kernel_launch(void* const* d_in, const int* in_sizes, int n_in,
                              void* d_out, int out_size, void* d_ws, size_t ws_size,
                              hipStream_t stream) {
    const float* x  = (const float*)d_in[0];
    const float* Wx = (const float*)d_in[1];
    const float* bx = (const float*)d_in[2];
    const float* Wh = (const float*)d_in[3];
    const float* bh = (const float*)d_in[4];
    const float* Wp = (const float*)d_in[5];
    const float* bp = (const float*)d_in[6];
    float* out = (float*)d_out;

    // Workspace carve-out (~87 MB total).
    uint8_t* p = (uint8_t*)d_ws;
    auto take = [&](size_t bytes) -> uint8_t* {
        uint8_t* r = p;
        p += (bytes + 255) & ~(size_t)255;
        return r;
    };
    _Float16* x16  = (_Float16*)take((size_t)B * S * D * sizeof(_Float16)); // 64 MB
    _Float16* Wcat = (_Float16*)take((size_t)NG * KC * sizeof(_Float16));   // 16 MB
    _Float16* Wp16 = (_Float16*)take((size_t)O * H * sizeof(_Float16));     //  2 MB
    float*    bsum = (float*)take((size_t)NG * sizeof(float));
    float*    cbuf = (float*)take((size_t)B * H * sizeof(float));
    float*    h32a = (float*)take((size_t)B * H * sizeof(float));
    float*    h32b = (float*)take((size_t)B * H * sizeof(float));
    _Float16* h16a = (_Float16*)take((size_t)B * H * sizeof(_Float16));
    _Float16* h16b = (_Float16*)take((size_t)B * H * sizeof(_Float16));
    unsigned* bar_cnt = (unsigned*)take(sizeof(unsigned));
    unsigned* bar_gen = (unsigned*)take(sizeof(unsigned));

    // Setup: f16 staging + combined bias + zero state + barrier re-arm.
    k_convert_x <<<(B * S * D) / (256 * 4), 256, 0, stream>>>(x, x16);
    k_build_wcat<<<((size_t)NG * KC) / 256, 256, 0, stream>>>(Wx, Wh, bx, bh, Wcat, bsum);
    k_build_wp  <<<((size_t)O * H) / 256, 256, 0, stream>>>(Wp, Wp16);
    k_init_state<<<(B * H + 255) / 256, 256, 0, stream>>>(cbuf, h32a, h32b, h16a, h16b,
                                                          bar_cnt, bar_gen, B * H);

    // Entire 512-step recurrence + all projections: ONE persistent launch.
    k_lstm_persistent<<<NBLOCKS, 256, 0, stream>>>(x16, Wcat, Wp16, bsum, bp, cbuf,
                                                   h16a, h16b, h32a, h32b, out,
                                                   bar_cnt, bar_gen);

    // Tail: final (c, h) into d_out after the 64x512x1024 outputs.
    k_tail<<<(B * H + 255) / 256, 256, 0, stream>>>(cbuf, h32a,
                                                    out + (size_t)B * S * O, B * H);
}
